// MSPartGuidedModule_16896401343296
// MI455X (gfx1250) — matrix operand order repaired and checked by reading, not compiled, not run on hardware
//
#include <hip/hip_runtime.h>
#include <hip/hip_bf16.h>

typedef __bf16 v16bf __attribute__((ext_vector_type(16)));
typedef float  v8f   __attribute__((ext_vector_type(8)));

static __device__ __forceinline__ unsigned short f2bf(float f) {
    // round-to-nearest-even float -> bf16 bits
    unsigned u = __float_as_uint(f);
    unsigned r = (u + 0x7FFFu + ((u >> 16) & 1u)) >> 16;
    return (unsigned short)r;
}

// ---------------------------------------------------------------------------
// Kernel 1: pack conv_w (2048x2048 f32, row-major [o][c]) into bf16 WMMA
// A-fragment layout: Wp[((o_tile*64 + kblk)*32 + lane)*16 + e]
//   lane = (o%16) + 16*(ko>>4),  e = ko&15,  ko = c%32
// ---------------------------------------------------------------------------
__global__ __launch_bounds__(256) void pack_w_kernel(const float* __restrict__ W,
                                                     unsigned short* __restrict__ Wp) {
    int idx = blockIdx.x * 256 + threadIdx.x;      // 0 .. 4194303
    int o = idx >> 11;
    int c = idx & 2047;
    int ot = o >> 4, r = o & 15;
    int kb = c >> 5, ko = c & 31;
    int lane = r + ((ko >> 4) << 4);
    int e    = ko & 15;
    size_t dst = ((size_t)(ot * 64 + kb) * 32 + lane) * 16 + e;
    Wp[dst] = f2bf(W[idx]);
}

// ---------------------------------------------------------------------------
// Kernel 2 (fused): per-(b,c) plane -> multiscale X, class sums, class means,
// and write Y = feats + classmean(label) as bf16 into K-transposed layout:
//   Yb[((b*128 + c/16)*512 + i)*16 + (c%16)]
// One wave per plane, 16 waves (16 channels) per block, B*128 blocks.
// ---------------------------------------------------------------------------
__global__ __launch_bounds__(512) void stats_pack_kernel(const float* __restrict__ feats,
                                                         const int* __restrict__ label,
                                                         unsigned short* __restrict__ Yb) {
    __shared__ float plane[16][512];                  // 32 KB
    __shared__ __align__(16) unsigned short ystage[512 * 16]; // 16 KB, [i][kk]
    __shared__ int   labsh[512];
    __shared__ int   cnt[6];
    __shared__ float pool4[16][32];
    __shared__ float pool8[16][8];
    __shared__ float pool16[16][2];

    const int tid  = threadIdx.x;
    const int wave = tid >> 5;
    const int lane = tid & 31;
    const int bc   = blockIdx.x;        // b*128 + c16
    const int b    = bc >> 7;
    const int c16  = bc & 127;

    if (tid < 6) cnt[tid] = 0;
    __syncthreads();

    int lb0 = label[b * 512 + tid];
    lb0 = lb0 < 0 ? 0 : (lb0 > 5 ? 5 : lb0);
    labsh[tid] = lb0;
    atomicAdd(&cnt[lb0], 1);

    const int c = c16 * 16 + wave;
    const float* fp = feats + ((size_t)b * 2048 + c) * 512;
    #pragma unroll
    for (int r = 0; r < 16; ++r)
        plane[wave][lane + 32 * r] = fp[lane + 32 * r];
    __syncthreads();

    // avgpool s=4 : 8x4 pools, one per lane (exact mean of 4x4 block)
    {
        int by = lane >> 2, bx = lane & 3;
        float s = 0.0f;
        #pragma unroll
        for (int dy = 0; dy < 4; ++dy)
            #pragma unroll
            for (int dx = 0; dx < 4; ++dx)
                s += plane[wave][(by * 4 + dy) * 16 + (bx * 4 + dx)];
        pool4[wave][lane] = s * (1.0f / 16.0f);
    }
    __syncthreads();
    if (lane < 8) {   // s=8 : 4x2, exact mean of four 4x4 pools
        int by = lane >> 1, bx = lane & 1;
        pool8[wave][lane] = 0.25f * (pool4[wave][(2 * by) * 4 + 2 * bx]     +
                                     pool4[wave][(2 * by) * 4 + 2 * bx + 1] +
                                     pool4[wave][(2 * by + 1) * 4 + 2 * bx] +
                                     pool4[wave][(2 * by + 1) * 4 + 2 * bx + 1]);
    }
    __syncthreads();
    if (lane < 2) {   // s=16 : 2x1
        pool16[wave][lane] = 0.25f * (pool8[wave][(2 * lane) * 2]     +
                                      pool8[wave][(2 * lane) * 2 + 1] +
                                      pool8[wave][(2 * lane + 1) * 2] +
                                      pool8[wave][(2 * lane + 1) * 2 + 1]);
    }
    __syncthreads();

    float S[6] = {0, 0, 0, 0, 0, 0};
    float fv[16];
    #pragma unroll
    for (int r = 0; r < 16; ++r) {
        const int p = lane + 32 * r;
        const int y = p >> 4, x = p & 15;
        const float f = plane[wave][p];
        fv[r] = f;
        // bilinear align_corners upsample of 8x4 pools
        float fy = y * (7.0f / 31.0f); int y0 = (int)fy; float wy = fy - y0; int y1 = y0 + 1 > 7 ? 7 : y0 + 1;
        float fx = x * (3.0f / 15.0f); int x0 = (int)fx; float wx = fx - x0; int x1 = x0 + 1 > 3 ? 3 : x0 + 1;
        float u4 = (pool4[wave][y0 * 4 + x0] * (1.0f - wx) + pool4[wave][y0 * 4 + x1] * wx) * (1.0f - wy)
                 + (pool4[wave][y1 * 4 + x0] * (1.0f - wx) + pool4[wave][y1 * 4 + x1] * wx) * wy;
        // 4x2 pools
        float fy8 = y * (3.0f / 31.0f); int a0 = (int)fy8; float wy8 = fy8 - a0; int a1 = a0 + 1 > 3 ? 3 : a0 + 1;
        float fx8 = x * (1.0f / 15.0f); int b0 = (int)fx8; float wx8 = fx8 - b0; int b1 = b0 + 1 > 1 ? 1 : b0 + 1;
        float u8 = (pool8[wave][a0 * 2 + b0] * (1.0f - wx8) + pool8[wave][a0 * 2 + b1] * wx8) * (1.0f - wy8)
                 + (pool8[wave][a1 * 2 + b0] * (1.0f - wx8) + pool8[wave][a1 * 2 + b1] * wx8) * wy8;
        // 2x1 pools (single column)
        float fy16 = y * (1.0f / 31.0f); int g0 = (int)fy16; float wy16 = fy16 - g0; int g1 = g0 + 1 > 1 ? 1 : g0 + 1;
        float u16 = pool16[wave][g0] * (1.0f - wy16) + pool16[wave][g1] * wy16;

        const float X = f + u4 + u8 + u16;
        const int lb = labsh[p];
        #pragma unroll
        for (int k = 0; k < 6; ++k) S[k] += (lb == k) ? X : 0.0f;
    }
    // butterfly reduce across wave -> every lane holds full sums (deterministic)
    #pragma unroll
    for (int k = 0; k < 6; ++k)
        #pragma unroll
        for (int off = 16; off > 0; off >>= 1)
            S[k] += __shfl_xor(S[k], off, 32);

    float M[6];
    M[0] = (S[0] + S[1] + S[2] + S[3] + S[4] + S[5]) * (1.0f / 512.0f); // label 0 -> global mean
    #pragma unroll
    for (int k = 1; k < 6; ++k) {
        int n = cnt[k];
        M[k] = S[k] / (float)(n > 0 ? n : 1);
    }

    #pragma unroll
    for (int r = 0; r < 16; ++r) {
        const int p = lane + 32 * r;
        const int lb = labsh[p];
        float m = M[0];
        #pragma unroll
        for (int k = 1; k < 6; ++k) m = (lb == k) ? M[k] : m;
        ystage[p * 16 + wave] = f2bf(fv[r] + m);   // Y = feats + context
    }
    __syncthreads();

    // cooperative coalesced store of the 16 KB [512][16] bf16 block
    uint4* dst = (uint4*)(Yb + (size_t)bc * 8192);
    const uint4* src = (const uint4*)ystage;
    #pragma unroll
    for (int k = tid; k < 1024; k += 512) dst[k] = src[k];
}

// ---------------------------------------------------------------------------
// Kernel 3: GEMM  z[b,o,i] = sum_c W[o,c] * Y[b,c,i]   via v_wmma_f32_16x16x32_bf16
// Per wave: 4 o-tiles x 2 i-tiles, K loop of 64 steps of 32.
// jobs = 16 (b) * 32 (o groups) * 16 (i groups) = 8192 waves
// ---------------------------------------------------------------------------
__global__ __launch_bounds__(256) void gemm_kernel(const unsigned short* __restrict__ Wp,
                                                   const unsigned short* __restrict__ Yb,
                                                   float* __restrict__ Z) {
    const int wave = threadIdx.x >> 5;
    const int lane = threadIdx.x & 31;
    const int job  = blockIdx.x * 8 + wave;
    const int b    = job >> 9;
    const int rem  = job & 511;
    const int ot   = rem >> 4;     // o-tile group (4 tiles of 16 -> 64 rows)
    const int it   = rem & 15;     // i-tile group (2 tiles of 16 -> 32 cols)
    const int laneLo = lane & 15;
    const int laneHi = lane >> 4;

    v8f acc[4][2];
    #pragma unroll
    for (int mt = 0; mt < 4; ++mt)
        #pragma unroll
        for (int nt = 0; nt < 2; ++nt)
            acc[mt][nt] = (v8f)0.0f;

    // B fragment base: Yb[((b*128 + kb*2 + laneHi)*512 + it*32 + laneLo)*16]
    const unsigned short* ybase =
        Yb + ((size_t)b * 128 * 512 + (size_t)laneHi * 512 + (size_t)(it * 2) * 16 + laneLo) * 16;
    // A fragment base: Wp[(((ot*4)*64 + kb)*32 + lane)*16]
    const unsigned short* abase = Wp + ((size_t)(ot * 4) * 2048 + lane) * 16;

    #pragma unroll 2
    for (int kb = 0; kb < 64; ++kb) {
        v16bf bf0 = *(const v16bf*)(ybase + (size_t)kb * 16384);
        v16bf bf1 = *(const v16bf*)(ybase + (size_t)kb * 16384 + 256);
        v16bf af[4];
        #pragma unroll
        for (int mt = 0; mt < 4; ++mt)
            af[mt] = *(const v16bf*)(abase + (size_t)mt * 32768 + (size_t)kb * 512);
        #pragma unroll
        for (int mt = 0; mt < 4; ++mt) {
            acc[mt][0] = __builtin_amdgcn_wmma_f32_16x16x32_bf16(
                false, af[mt], false, bf0, (short)0, acc[mt][0], false, false);
            acc[mt][1] = __builtin_amdgcn_wmma_f32_16x16x32_bf16(
                false, af[mt], false, bf1, (short)0, acc[mt][1], false, false);
        }
    }

    // D layout: lanes 0-15 hold N=lane, rows M=r ; lanes 16-31 hold N=lane-16, rows M=r+8
    #pragma unroll
    for (int mt = 0; mt < 4; ++mt) {
        #pragma unroll
        for (int nt = 0; nt < 2; ++nt) {
            const int i = (it * 2 + nt) * 16 + laneLo;
            float* zp = Z + ((size_t)b * 2048 + (size_t)((ot * 4 + mt) * 16 + laneHi * 8)) * 512 + i;
            #pragma unroll
            for (int r = 0; r < 8; ++r)
                zp[(size_t)r * 512] = acc[mt][nt][r];
        }
    }
}

// ---------------------------------------------------------------------------
// Kernel 4: per-channel BN statistics over (B, HW) = 8192 values (deterministic)
// ---------------------------------------------------------------------------
__global__ __launch_bounds__(256) void bn_stats_kernel(const float* __restrict__ Z,
                                                       float* __restrict__ meanArr,
                                                       float* __restrict__ rstdArr) {
    const int o = blockIdx.x;
    float s = 0.0f, s2 = 0.0f;
    for (int b = 0; b < 16; ++b) {
        const float* zp = Z + ((size_t)b * 2048 + o) * 512;
        #pragma unroll
        for (int i = threadIdx.x; i < 512; i += 256) {
            float v = zp[i];
            s += v; s2 += v * v;
        }
    }
    __shared__ float sh0[256];
    __shared__ float sh1[256];
    sh0[threadIdx.x] = s; sh1[threadIdx.x] = s2;
    __syncthreads();
    for (int off = 128; off > 0; off >>= 1) {
        if (threadIdx.x < off) {
            sh0[threadIdx.x] += sh0[threadIdx.x + off];
            sh1[threadIdx.x] += sh1[threadIdx.x + off];
        }
        __syncthreads();
    }
    if (threadIdx.x == 0) {
        float mean = sh0[0] * (1.0f / 8192.0f);
        float var  = sh1[0] * (1.0f / 8192.0f) - mean * mean;
        meanArr[o] = mean;
        rstdArr[o] = rsqrtf(var + 1e-5f);
    }
}

// ---------------------------------------------------------------------------
// Kernel 5: apply BN affine
// ---------------------------------------------------------------------------
__global__ __launch_bounds__(256) void bn_apply_kernel(const float* __restrict__ Z,
                                                       const float* __restrict__ meanArr,
                                                       const float* __restrict__ rstdArr,
                                                       const float* __restrict__ gamma,
                                                       const float* __restrict__ beta,
                                                       float* __restrict__ out) {
    size_t idx = (size_t)blockIdx.x * 256 + threadIdx.x;   // 16,777,216 total
    int o = (int)((idx >> 9) & 2047);
    out[idx] = (Z[idx] - meanArr[o]) * rstdArr[o] * gamma[o] + beta[o];
}

// ---------------------------------------------------------------------------
extern "C" void kernel_launch(void* const* d_in, const int* in_sizes, int n_in,
                              void* d_out, int out_size, void* d_ws, size_t ws_size,
                              hipStream_t stream) {
    (void)in_sizes; (void)n_in; (void)out_size; (void)ws_size;
    const float* feats  = (const float*)d_in[0];   // (16, 2048, 32, 16)
    const int*   label  = (const int*)d_in[1];     // (16, 32, 16)
    const float* conv_w = (const float*)d_in[2];   // (2048, 2048)
    const float* gamma  = (const float*)d_in[3];   // (2048,)
    const float* beta   = (const float*)d_in[4];   // (2048,)
    float* out = (float*)d_out;

    char* ws = (char*)d_ws;
    unsigned short* Wp = (unsigned short*)ws;                               //  8 MB
    unsigned short* Yb = (unsigned short*)(ws + (8ull << 20));              // 32 MB
    float* Z           = (float*)(ws + (40ull << 20));                      // 64 MB
    float* meanArr     = (float*)(ws + (104ull << 20));                     //  8 KB
    float* rstdArr     = meanArr + 2048;

    pack_w_kernel   <<<16384, 256, 0, stream>>>(conv_w, Wp);
    stats_pack_kernel<<<2048, 512, 0, stream>>>(feats, label, Yb);
    gemm_kernel     <<<1024, 256, 0, stream>>>(Wp, Yb, Z);
    bn_stats_kernel <<<2048, 256, 0, stream>>>(Z, meanArr, rstdArr);
    bn_apply_kernel <<<65536, 256, 0, stream>>>(Z, meanArr, rstdArr, gamma, beta, out);
}
